// BasicBlock_37391985279093
// MI455X (gfx1250) — matrix operand order repaired and checked by reading, not compiled, not run on hardware
//
#include <hip/hip_runtime.h>

typedef __attribute__((ext_vector_type(16))) __bf16 v16bf;
typedef __attribute__((ext_vector_type(8)))  __bf16 v8bf;
typedef __attribute__((ext_vector_type(8)))  float  v8f;

// Fixed problem shape
#define TT        4
#define BBATCH    32
#define CCH       64
#define HH        56
#define WWID      56
#define HWPIX     3136        // 56*56
#define NIMG      128         // T*B
#define NPIXTOT   401408      // NIMG*HW
#define NTOT      25690112    // NIMG*HW*C
#define KTOT      576         // C*9
#define PER_T     6422528     // B*HW*C  (== B*C*HW)
#define STAT_BLOCKS 1024

// ---------- helpers ----------
__device__ __forceinline__ unsigned short f32_to_bf16_rne(float f) {
    union { float f; unsigned u; } v; v.f = f;
    unsigned u = v.u;
    unsigned r = u + 0x7FFFu + ((u >> 16) & 1u);
    if ((u & 0x7F800000u) == 0x7F800000u) r = u;   // Inf/NaN: truncate
    return (unsigned short)(r >> 16);
}

__device__ __forceinline__ v16bf ld_frag(const __bf16* p0, const __bf16* p1) {
    v8bf lo = *(const v8bf*)p0;
    v8bf hi = *(const v8bf*)p1;
    return __builtin_shufflevector(lo, hi, 0,1,2,3,4,5,6,7,8,9,10,11,12,13,14,15);
}

// ---------- pack x: f32 NCHW -> bf16 NHWC ----------
__global__ __launch_bounds__(256)
void pack_x_nhwc_bf16(const float* __restrict__ x, unsigned short* __restrict__ xb) {
    long j = (long)blockIdx.x * 256 + threadIdx.x;          // output-major index
    if (j >= (long)NTOT) return;
    int  c   = (int)(j & 63);
    long rem = j >> 6;
    int  hw  = (int)(rem % HWPIX);
    long n   = rem / HWPIX;
    float v = x[(n * CCH + c) * (long)HWPIX + hw];
    xb[j] = f32_to_bf16_rne(v);
}

// ---------- pack conv weight: [co][ci][3][3] f32 -> [co][K=tap*64+ci] bf16 ----------
__global__ __launch_bounds__(256)
void pack_w_bf16(const float* __restrict__ w, unsigned short* __restrict__ wp) {
    int j = blockIdx.x * 256 + threadIdx.x;                 // input-major
    if (j >= CCH * KTOT) return;
    int co  = j / KTOT;
    int r   = j % KTOT;
    int ci  = r / 9;
    int tap = r % 9;
    wp[co * KTOT + tap * CCH + ci] = f32_to_bf16_rne(w[j]);
}

// ---------- implicit-GEMM 3x3 conv, bf16 WMMA, NHWC ----------
// act: bf16 [NIMG,56,56,64]; wgt: bf16 [64 co][576 k]; out: f32 [NIMG,56,56,64]
// Block: 256 thr (8 waves) -> 128 pixels x 64 co.
// Wave tile: 32 px x 32 co = 4 accumulators; 4 WMMAs per K-chunk, A/B frags reused 2x each.
__global__ __launch_bounds__(256)
void conv3x3_wmma_bf16(const __bf16* __restrict__ act,
                       const __bf16* __restrict__ wgt,
                       float* __restrict__ out) {
    __shared__ __bf16 sW[CCH * KTOT];       // full weight matrix: 72 KB (of 320 KB WGP LDS)
    __shared__ __bf16 sA[2][128 * 32];      // double-buffered 128-pixel x 32-K act tiles (2x8 KB)

    const int t      = threadIdx.x;
    const int lane   = t & 31;
    const int waveId = t >> 5;
    const int half   = lane >> 4;
    const int lr     = lane & 15;

    // preload entire weight matrix into LDS (144 bf16 per thread, 16B vectors)
    #pragma unroll
    for (int i = 0; i < 18; ++i) {
        int e = (i * 256 + t) * 8;
        *(v8bf*)&sW[e] = *(const v8bf*)&wgt[e];
    }

    // staging geometry: thread t stages pixel t>>1, 16 channels at (t&1)*16
    const long pixBlock = (long)blockIdx.x * 128;
    const int  stPix    = t >> 1;
    const int  stC      = (t & 1) * 16;
    const long gp       = pixBlock + stPix;
    const int  n        = (int)(gp / HWPIX);
    const int  rem      = (int)(gp % HWPIX);
    const int  py       = rem / WWID;
    const int  px       = rem % WWID;

    const int pixBase = (waveId >> 1) * 32;   // 0,32,64,96
    const int coBase  = (waveId & 1) * 32;    // 0,32

    v8f acc00 = {}, acc01 = {}, acc10 = {}, acc11 = {};

    // stage K-chunk 0: tap 0 (dy=-1,dx=-1), ci0=0
    {
        int yy = py - 1, xx = px - 1;
        ulonglong2 v0 = {0ull, 0ull}, v1 = {0ull, 0ull};
        if (yy >= 0 && xx >= 0) {
            const ulonglong2* src =
                (const ulonglong2*)&act[(((long)n * HWPIX + yy * WWID + xx) * CCH) + stC];
            v0 = src[0]; v1 = src[1];
        }
        ulonglong2* dst = (ulonglong2*)&sA[0][stPix * 32 + stC];
        dst[0] = v0; dst[1] = v1;
    }
    __syncthreads();

    for (int kc = 0; kc < 18; ++kc) {
        // stage next chunk into the other buffer
        if (kc + 1 < 18) {
            int kn  = kc + 1;
            int tap = kn >> 1;
            int dy  = tap / 3 - 1;
            int dx  = tap % 3 - 1;
            int ci0 = (kn & 1) * 32;
            int yy = py + dy, xx = px + dx;
            ulonglong2 v0 = {0ull, 0ull}, v1 = {0ull, 0ull};
            if (yy >= 0 && yy < HH && xx >= 0 && xx < WWID) {
                const ulonglong2* src =
                    (const ulonglong2*)&act[(((long)n * HWPIX + yy * WWID + xx) * CCH) + ci0 + stC];
                v0 = src[0]; v1 = src[1];
            }
            ulonglong2* dst = (ulonglong2*)&sA[kn & 1][stPix * 32 + stC];
            dst[0] = v0; dst[1] = v1;
        }

        // A fragments (16 px x 32 K each), ISA 16-bit A layout:
        //   e<8 -> K = e + 8*half ; e>=8 -> K = e + 8 + 8*half  => two contiguous 8-elem runs
        const __bf16* base = sA[kc & 1];
        const __bf16* ar0 = &base[(pixBase + lr) * 32];
        const __bf16* ar1 = &base[(pixBase + 16 + lr) * 32];
        v16bf a0 = ld_frag(&ar0[half * 8], &ar0[16 + half * 8]);
        v16bf a1 = ld_frag(&ar1[half * 8], &ar1[16 + half * 8]);

        // B fragments (32 K x 16 co each), same K order as A: e<8 -> K = e + 8*half ; e>=8 -> K = e + 8 + 8*half
        const __bf16* wb0 = &sW[(coBase + lr) * KTOT + kc * 32 + 8 * half];
        const __bf16* wb1 = &sW[(coBase + 16 + lr) * KTOT + kc * 32 + 8 * half];
        v16bf b0 = ld_frag(wb0, wb0 + 16);
        v16bf b1 = ld_frag(wb1, wb1 + 16);

        acc00 = __builtin_amdgcn_wmma_f32_16x16x32_bf16(false, a0, false, b0, (short)0, acc00, false, false);
        acc01 = __builtin_amdgcn_wmma_f32_16x16x32_bf16(false, a0, false, b1, (short)0, acc01, false, false);
        acc10 = __builtin_amdgcn_wmma_f32_16x16x32_bf16(false, a1, false, b0, (short)0, acc10, false, false);
        acc11 = __builtin_amdgcn_wmma_f32_16x16x32_bf16(false, a1, false, b1, (short)0, acc11, false, false);
        __syncthreads();
    }

    // D layout: lane col = co, VGPR i -> row M = i + 8*half
    const int co0 = coBase + lr;
    const int co1 = coBase + 16 + lr;
    #pragma unroll
    for (int i = 0; i < 8; ++i) {
        long p0 = pixBlock + pixBase + i + 8 * half;
        long p1 = p0 + 16;
        out[p0 * CCH + co0] = acc00[i];
        out[p0 * CCH + co1] = acc01[i];
        out[p1 * CCH + co0] = acc10[i];
        out[p1 * CCH + co1] = acc11[i];
    }
}

// ---------- deterministic per-channel stats: stage 1 (per-block partials) ----------
__global__ __launch_bounds__(256)
void stats_partial(const float* __restrict__ y, float* __restrict__ partial) {
    __shared__ float ssum[4][64];
    __shared__ float ssq[4][64];
    const int c  = threadIdx.x & 63;
    const int pr = threadIdx.x >> 6;
    float s = 0.f, q = 0.f;
    for (long p = (long)blockIdx.x * 4 + pr; p < (long)NPIXTOT; p += 4L * STAT_BLOCKS) {
        float v = y[p * CCH + c];
        s += v; q += v * v;
    }
    ssum[pr][c] = s; ssq[pr][c] = q;
    __syncthreads();
    if (pr == 0) {
        float S = ssum[0][c] + ssum[1][c] + ssum[2][c] + ssum[3][c];
        float Q = ssq[0][c]  + ssq[1][c]  + ssq[2][c]  + ssq[3][c];
        partial[((long)blockIdx.x * 64 + c) * 2 + 0] = S;
        partial[((long)blockIdx.x * 64 + c) * 2 + 1] = Q;
    }
}

// ---------- stats stage 2: fixed-order final sum -> mean, rstd ----------
__global__ void stats_final(const float* __restrict__ partial, float* __restrict__ stats) {
    int c = threadIdx.x;
    if (c >= 64) return;
    float S = 0.f, Q = 0.f;
    for (int b = 0; b < STAT_BLOCKS; ++b) {
        S += partial[((long)b * 64 + c) * 2 + 0];
        Q += partial[((long)b * 64 + c) * 2 + 1];
    }
    float mean = S / (float)NPIXTOT;
    float var  = Q / (float)NPIXTOT - mean * mean;
    stats[c * 2 + 0] = mean;
    stats[c * 2 + 1] = rsqrtf(var + 1e-5f);
}

// ---------- fused BN + PLIF -> bf16 spikes (conv2 input) ----------
__global__ __launch_bounds__(256)
void bn_plif_spike(const float* __restrict__ y,       // [T*B, HW, C] f32
                   const float* __restrict__ stats,   // [C][mean,rstd]
                   const float* __restrict__ gamma,
                   const float* __restrict__ beta,
                   const float* __restrict__ lifw,
                   unsigned short* __restrict__ spk)  // bf16 [T*B, HW, C]
{
    long j = (long)blockIdx.x * 256 + threadIdx.x;
    if (j >= (long)PER_T) return;
    const int   c    = (int)(j & 63);
    const float mean = stats[c * 2], rstd = stats[c * 2 + 1];
    const float g = gamma[c], bt = beta[c];
    const float sig = 1.f / (1.f + __expf(-lifw[0]));
    float v = 0.f;
    #pragma unroll
    for (int t = 0; t < TT; ++t) {
        float xv = y[(long)t * PER_T + j];
        xv = (xv - mean) * rstd * g + bt;
        v = v + (xv - v) * sig;
        float s = (v - 1.f >= 0.f) ? 1.f : 0.f;
        v = v * (1.f - s);
        spk[(long)t * PER_T + j] = (s != 0.f) ? (unsigned short)0x3F80 : (unsigned short)0;
    }
}

// ---------- final: residual + BN2 + PLIF -> f32 NCHW output ----------
__global__ __launch_bounds__(256)
void residual_bn_plif_out(const float* __restrict__ x,     // [T,B,C,HW] f32
                          const float* __restrict__ y2,    // [T*B, HW, C] f32
                          const float* __restrict__ stats,
                          const float* __restrict__ gamma,
                          const float* __restrict__ beta,
                          const float* __restrict__ lifw,
                          float* __restrict__ out)         // [T,B,C,HW] f32
{
    long j = (long)blockIdx.x * 256 + threadIdx.x;         // index within one time slice, NCHW
    if (j >= (long)PER_T) return;
    const int  hw = (int)(j % HWPIX);
    const long bc = j / HWPIX;
    const int  c  = (int)(bc & 63);
    const long b  = bc >> 6;
    const float mean = stats[c * 2], rstd = stats[c * 2 + 1];
    const float g = gamma[c], bt = beta[c];
    const float sig = 1.f / (1.f + __expf(-lifw[0]));
    const long nhwcBase = (b * HWPIX + hw) * CCH + c;
    float v = 0.f;
    #pragma unroll
    for (int t = 0; t < TT; ++t) {
        float h = y2[(long)t * BBATCH * HWPIX * CCH + nhwcBase];
        h = (h - mean) * rstd * g + bt;
        float u = x[(long)t * PER_T + j] + h;
        v = v + (u - v) * sig;
        float s = (v - 1.f >= 0.f) ? 1.f : 0.f;
        v = v * (1.f - s);
        out[(long)t * PER_T + j] = s;
    }
}

// ---------- launch ----------
extern "C" void kernel_launch(void* const* d_in, const int* in_sizes, int n_in,
                              void* d_out, int out_size, void* d_ws, size_t ws_size,
                              hipStream_t stream) {
    (void)in_sizes; (void)n_in; (void)out_size; (void)ws_size;

    const float* x       = (const float*)d_in[0];
    const float* conv1_w = (const float*)d_in[1];
    const float* bn1_g   = (const float*)d_in[2];
    const float* bn1_b   = (const float*)d_in[3];
    const float* lif1_w  = (const float*)d_in[4];
    const float* conv2_w = (const float*)d_in[5];
    const float* bn2_g   = (const float*)d_in[6];
    const float* bn2_b   = (const float*)d_in[7];
    const float* lif2_w  = (const float*)d_in[8];
    float* out = (float*)d_out;

    // workspace carve-up (256B aligned)
    char* ws = (char*)d_ws;
    size_t off = 0;
    auto carve = [&](size_t bytes) { char* p = ws + off; off = (off + bytes + 255) & ~(size_t)255; return p; };
    unsigned short* bufA   = (unsigned short*)carve((size_t)NTOT * 2);   // x-pack, later spikes
    float*          bufB   = (float*)carve((size_t)NTOT * 4);            // conv1 out, later conv2 out
    unsigned short* w1p    = (unsigned short*)carve((size_t)CCH * KTOT * 2);
    unsigned short* w2p    = (unsigned short*)carve((size_t)CCH * KTOT * 2);
    float*          part   = (float*)carve((size_t)STAT_BLOCKS * 64 * 2 * 4);
    float*          stats1 = (float*)carve(64 * 2 * 4);
    float*          stats2 = (float*)carve(64 * 2 * 4);

    const int convBlocks = NPIXTOT / 128;  // 3136
    const int perTBlocks = PER_T / 256;    // 25088

    // 0) packs
    pack_x_nhwc_bf16<<<NTOT / 256, 256, 0, stream>>>(x, bufA);
    pack_w_bf16<<<(CCH * KTOT + 255) / 256, 256, 0, stream>>>(conv1_w, w1p);
    pack_w_bf16<<<(CCH * KTOT + 255) / 256, 256, 0, stream>>>(conv2_w, w2p);

    // 1) conv1 (WMMA implicit GEMM)
    conv3x3_wmma_bf16<<<convBlocks, 256, 0, stream>>>((const __bf16*)bufA, (const __bf16*)w1p, bufB);

    // 2) BN1 stats (deterministic two-stage)
    stats_partial<<<STAT_BLOCKS, 256, 0, stream>>>(bufB, part);
    stats_final<<<1, 64, 0, stream>>>(part, stats1);

    // 3) BN1 + PLIF1 -> bf16 spikes (reuse bufA)
    bn_plif_spike<<<perTBlocks, 256, 0, stream>>>(bufB, stats1, bn1_g, bn1_b, lif1_w, bufA);

    // 4) conv2 on spikes (reuse bufB)
    conv3x3_wmma_bf16<<<convBlocks, 256, 0, stream>>>((const __bf16*)bufA, (const __bf16*)w2p, bufB);

    // 5) BN2 stats
    stats_partial<<<STAT_BLOCKS, 256, 0, stream>>>(bufB, part);
    stats_final<<<1, 64, 0, stream>>>(part, stats2);

    // 6) residual + BN2 + PLIF2 -> NCHW f32 output
    residual_bn_plif_out<<<perTBlocks, 256, 0, stream>>>(x, bufB, stats2, bn2_g, bn2_b, lif2_w, out);
}
